// Model_52613349376643
// MI455X (gfx1250) — matrix operand order repaired and checked by reading, not compiled, not run on hardware
//
#include <hip/hip_runtime.h>
#include <math.h>

// ---------------------------------------------------------------------------
// Autoformer forward (CDNA5 / gfx1250, wave32).
// GEMMs + circular-conv embeddings run on v_wmma_f32_16x16x32_bf16 with
// pre-packed bf16 operands (no cvt in the hot loop) and 64x32 register tiles
// per wave (8 WMMAs per K-chunk).  Autocorrelation = direct circular
// cross-correlation averaged over channels (== irfft(rfft(q)*conj(rfft(k)))
// meaned over (H,E)), so no FFT is needed.
// ---------------------------------------------------------------------------

#define BATCH   32
#define D_MODEL 512
#define D_FF    1024
#define L_ENC   512
#define L_DEC   768   // LABEL_LEN + PRED_LEN
#define LABEL   256
#define MA_K    25
#define MA_PAD  12
#define EPSV    1e-5f

typedef __attribute__((ext_vector_type(16))) __bf16 v16bf;
typedef __attribute__((ext_vector_type(8)))  __bf16 v8bf;
typedef __attribute__((ext_vector_type(8)))  float  v8f;

#define TM 4   // 16-row tiles per wave  (64 rows)
#define TN 2   // 16-col tiles per wave  (32 cols)

// ---------------------------------------------------------------------------
// fp32 -> bf16 streaming pack
// ---------------------------------------------------------------------------
__global__ void pack_bf16_kernel(const float* __restrict__ src,
                                 __bf16* __restrict__ dst, size_t n)
{
    const size_t idx = (size_t)blockIdx.x * blockDim.x + threadIdx.x;
    if (idx < n) dst[idx] = (__bf16)src[idx];
}

// conv weight [O,C,3] (stride-3) -> bf16 [3][O][C] contiguous
__global__ void pack_convw_kernel(const float* __restrict__ W,
                                  __bf16* __restrict__ dst, int O, int C)
{
    const size_t n = (size_t)3 * O * C;
    const size_t idx = (size_t)blockIdx.x * blockDim.x + threadIdx.x;
    if (idx >= n) return;
    const int i   = (int)(idx / ((size_t)O * C));
    const int rem = (int)(idx - (size_t)i * O * C);
    const int o = rem / C, c = rem - o * C;
    dst[idx] = (__bf16)W[(size_t)o * C * 3 + (size_t)c * 3 + i];
}

// ---------------------------------------------------------------------------
// WMMA GEMM:  C[M,N] = act(A[M,K] @ W[N,K]^T + bias) + res
// A, W pre-packed bf16 row-major.  Per-wave 64x32 output (4x2 16x16 tiles).
// Output to fp32 (Cf) or bf16 (Cb) -- exactly one non-null.
// A/B fragments: contiguous k order, k = 16*(lane/16) + s, in both.
// ---------------------------------------------------------------------------
__global__ void gemm_bf16_wmma_kernel(const __bf16* __restrict__ A,
                                      const __bf16* __restrict__ W,
                                      const float* __restrict__ bias,
                                      const float* __restrict__ res,
                                      float* __restrict__ Cf,
                                      __bf16* __restrict__ Cb,
                                      int M, int N, int K, int act)
{
    const int tileN0 = blockIdx.x * TN;
    const int tileM0 = (blockIdx.y * blockDim.y + threadIdx.y) * TM;
    if (tileM0 * 16 >= M) return;          // uniform per wave

    const int lane = threadIdx.x;
    const int r16  = lane & 15;
    const int hi   = lane >> 4;
    const int kbA  = hi * 16;              // lanes 16-31: +16
    const int kbB  = hi * 16;              // lanes 16-31: +16

    const __bf16* Arow[TM];
    const __bf16* Wrow[TN];
#pragma unroll
    for (int i = 0; i < TM; ++i)
        Arow[i] = A + (size_t)((tileM0 + i) * 16 + r16) * K;
#pragma unroll
    for (int j = 0; j < TN; ++j)
        Wrow[j] = W + (size_t)((tileN0 + j) * 16 + r16) * K;

    v8f acc[TM][TN] = {};

    for (int k0 = 0; k0 < K; k0 += 32) {
        v16bf a[TM], b[TN];
#pragma unroll
        for (int i = 0; i < TM; ++i) {
            const v8bf a0 = *(const v8bf*)(Arow[i] + k0 + kbA);
            const v8bf a1 = *(const v8bf*)(Arow[i] + k0 + kbA + 8);
            a[i] = __builtin_shufflevector(a0, a1, 0,1,2,3,4,5,6,7,
                                                   8,9,10,11,12,13,14,15);
        }
#pragma unroll
        for (int j = 0; j < TN; ++j)
            b[j] = *(const v16bf*)(Wrow[j] + k0 + kbB);
#pragma unroll
        for (int i = 0; i < TM; ++i)
#pragma unroll
            for (int j = 0; j < TN; ++j)
                acc[i][j] = __builtin_amdgcn_wmma_f32_16x16x32_bf16(
                                false, a[i], false, b[j],
                                (short)0, acc[i][j], false, false);
    }

#pragma unroll
    for (int j = 0; j < TN; ++j) {
        const int n = (tileN0 + j) * 16 + r16;
        const float bv = bias ? bias[n] : 0.0f;
#pragma unroll
        for (int i = 0; i < TM; ++i) {
#pragma unroll
            for (int rr = 0; rr < 8; ++rr) {
                const int m = (tileM0 + i) * 16 + rr + 8 * hi;
                float v = acc[i][j][rr] + bv;
                if (act == 1) v = 0.5f * v * (1.0f + erff(v * 0.70710678118654752f));
                const size_t idx = (size_t)m * N + n;
                if (res) v += res[idx];
                if (Cf) Cf[idx] = v; else Cb[idx] = (__bf16)v;
            }
        }
    }
}

// ---------------------------------------------------------------------------
// WMMA circular conv: out[b,l,o] = sum_{i<3,c} X[b,(l-1+i)%L,c]*W[o,c,i]
//                      (+ mark @ Wtemp^T) (+ res)
// X pre-packed bf16 [B,L,C]; W pre-packed bf16 [3][O][C].
// Per-wave 16x32 output tile (1x2).  L multiple of 16 -> tile within one b.
// ---------------------------------------------------------------------------
__global__ void circconv_bf16_wmma_kernel(const __bf16* __restrict__ X,
                                          const __bf16* __restrict__ Wp,
                                          const float* __restrict__ mark,
                                          const float* __restrict__ Wtemp,
                                          const float* __restrict__ res,
                                          float* __restrict__ C,
                                          int L, int Cin, int O)
{
    const int tileN0 = blockIdx.x * 2;
    const int tileM  = blockIdx.y * blockDim.y + threadIdx.y;
    if (tileM * 16 >= BATCH * L) return;

    const int lane = threadIdx.x;
    const int r16  = lane & 15;
    const int hi   = lane >> 4;
    const int kbA  = hi * 16;
    const int kbB  = hi * 16;

    const int m  = tileM * 16 + r16;
    const int bb = m / L;
    const int l  = m - bb * L;
    const __bf16* Xb = X + (size_t)bb * L * Cin;
    const __bf16* rows[3];
#pragma unroll
    for (int i = 0; i < 3; ++i) {
        int ls = l - 1 + i;
        if (ls < 0)  ls += L;
        if (ls >= L) ls -= L;
        rows[i] = Xb + (size_t)ls * Cin;
    }

    v8f acc[2] = {};

#pragma unroll
    for (int i = 0; i < 3; ++i) {
        const __bf16* Ar = rows[i];
        const __bf16* W0r = Wp + ((size_t)i * O + (tileN0 + 0) * 16 + r16) * Cin;
        const __bf16* W1r = Wp + ((size_t)i * O + (tileN0 + 1) * 16 + r16) * Cin;
        for (int k0 = 0; k0 < Cin; k0 += 32) {
            const v8bf a0 = *(const v8bf*)(Ar + k0 + kbA);
            const v8bf a1 = *(const v8bf*)(Ar + k0 + kbA + 8);
            const v16bf a = __builtin_shufflevector(a0, a1, 0,1,2,3,4,5,6,7,
                                                            8,9,10,11,12,13,14,15);
            const v16bf b0 = *(const v16bf*)(W0r + k0 + kbB);
            const v16bf b1 = *(const v16bf*)(W1r + k0 + kbB);
            acc[0] = __builtin_amdgcn_wmma_f32_16x16x32_bf16(false, a, false, b0,
                                                             (short)0, acc[0], false, false);
            acc[1] = __builtin_amdgcn_wmma_f32_16x16x32_bf16(false, a, false, b1,
                                                             (short)0, acc[1], false, false);
        }
    }

#pragma unroll
    for (int j = 0; j < 2; ++j) {
        const int n = (tileN0 + j) * 16 + r16;
#pragma unroll
        for (int rr = 0; rr < 8; ++rr) {
            const int m2 = tileM * 16 + rr + 8 * hi;
            float v = acc[j][rr];
            if (mark) {
                const float* mk = mark + (size_t)m2 * 4;   // [B,L,4] rows
                const float* wt = Wtemp + (size_t)n * 4;
                v += mk[0]*wt[0] + mk[1]*wt[1] + mk[2]*wt[2] + mk[3]*wt[3];
            }
            const size_t idx = (size_t)m2 * O + n;
            if (res) v += res[idx];
            C[idx] = v;
        }
    }
}

// ---------------------------------------------------------------------------
// series_decomp: moving average (k=25, edge-replicate pad) -> trend,
// seasonal = x - trend.  Either output may be null; trend may accumulate.
// ---------------------------------------------------------------------------
__global__ void decomp_kernel(const float* __restrict__ x,
                              float* __restrict__ seasonal,
                              float* __restrict__ trend,
                              int L, int C, int accum)
{
    const size_t n = (size_t)BATCH * L * C;
    const size_t idx = (size_t)blockIdx.x * blockDim.x + threadIdx.x;
    if (idx >= n) return;
    const int c = (int)(idx % C);
    const size_t bl = idx / C;
    const int l = (int)(bl % L);
    const int b = (int)(bl / L);
    const float* xb = x + ((size_t)b * L) * C + c;
    float s = 0.f;
#pragma unroll
    for (int j = 0; j < MA_K; ++j) {
        int ll = l + j - MA_PAD;
        ll = ll < 0 ? 0 : (ll >= L ? L - 1 : ll);
        s += xb[(size_t)ll * C];
    }
    const float mov = s * (1.0f / MA_K);
    if (seasonal) seasonal[idx] = x[idx] - mov;
    if (trend)    trend[idx] = accum ? trend[idx] + mov : mov;
}

// layernorm over last dim (biased var), *g + b.  One block per row.
__global__ void layernorm_kernel(const float* __restrict__ x,
                                 const float* __restrict__ g,
                                 const float* __restrict__ bta,
                                 float* __restrict__ y, int C)
{
    __shared__ float red[256];
    const size_t row = blockIdx.x;
    const float* xr = x + row * (size_t)C;
    float s = 0.f;
    for (int c = threadIdx.x; c < C; c += 256) s += xr[c];
    red[threadIdx.x] = s; __syncthreads();
    for (int o = 128; o > 0; o >>= 1) {
        if (threadIdx.x < o) red[threadIdx.x] += red[threadIdx.x + o];
        __syncthreads();
    }
    const float mu = red[0] / C;
    __syncthreads();
    float s2 = 0.f;
    for (int c = threadIdx.x; c < C; c += 256) { float d = xr[c] - mu; s2 += d * d; }
    red[threadIdx.x] = s2; __syncthreads();
    for (int o = 128; o > 0; o >>= 1) {
        if (threadIdx.x < o) red[threadIdx.x] += red[threadIdx.x + o];
        __syncthreads();
    }
    const float inv = rsqrtf(red[0] / C + EPSV);
    for (int c = threadIdx.x; c < C; c += 256)
        y[row * (size_t)C + c] = (xr[c] - mu) * inv * g[c] + bta[c];
}

// subtract per-(b,c) mean over time axis (xh - xh.mean(axis=1))
__global__ void subtimemean_kernel(float* __restrict__ y, int L, int C)
{
    const int idx = blockIdx.x * blockDim.x + threadIdx.x;
    if (idx >= BATCH * C) return;
    const int b = idx / C, c = idx % C;
    float* p = y + (size_t)b * L * C + c;
    float s = 0.f;
    for (int l = 0; l < L; ++l) s += p[(size_t)l * C];
    s /= L;
    for (int l = 0; l < L; ++l) p[(size_t)l * C] -= s;
}

// mean over time axis -> out[b,c]
__global__ void meantime_kernel(const float* __restrict__ x,
                                float* __restrict__ out, int L, int C)
{
    const int idx = blockIdx.x * blockDim.x + threadIdx.x;
    if (idx >= BATCH * C) return;
    const int b = idx / C, c = idx % C;
    const float* p = x + (size_t)b * L * C + c;
    float s = 0.f;
    for (int l = 0; l < L; ++l) s += p[(size_t)l * C];
    out[idx] = s / L;
}

// corr[b,tau] = (1/512) * sum_{l,d} q[b,(l+tau)%L,d] * k[b,l,d]
__global__ void meancorr_kernel(const float* __restrict__ q,
                                const float* __restrict__ k,
                                float* __restrict__ corr, int L)
{
    __shared__ float red[256];
    const int tau = blockIdx.x, b = blockIdx.y;
    const float* qb = q + (size_t)b * L * D_MODEL;
    const float* kb = k + (size_t)b * L * D_MODEL;
    float s = 0.f;
    const int n = L * D_MODEL;
    for (int i = threadIdx.x; i < n; i += 256) {
        const int l = i >> 9;            // D_MODEL == 512
        const int d = i & 511;
        int lq = l + tau; if (lq >= L) lq -= L;
        s += qb[(size_t)lq * D_MODEL + d] * kb[i];
    }
    red[threadIdx.x] = s; __syncthreads();
    for (int o = 128; o > 0; o >>= 1) {
        if (threadIdx.x < o) red[threadIdx.x] += red[threadIdx.x + o];
        __syncthreads();
    }
    if (threadIdx.x == 0) corr[b * L + tau] = red[0] * (1.0f / 512.0f);
}

// iterative top-k (max + mask) then softmax of the k values.  One block per b.
__global__ void topk_softmax_kernel(const float* __restrict__ corr,
                                    int* __restrict__ delays,
                                    float* __restrict__ wts, int L, int topk)
{
    __shared__ float vals[1024];
    __shared__ float rv[256];
    __shared__ int   ri[256];
    __shared__ float tv[32];
    __shared__ int   tix[32];
    const int b = blockIdx.x;
    for (int i = threadIdx.x; i < L; i += 256) vals[i] = corr[b * L + i];
    __syncthreads();
    for (int t = 0; t < topk; ++t) {
        float best = -3.4e38f; int bi = 0;
        for (int i = threadIdx.x; i < L; i += 256)
            if (vals[i] > best) { best = vals[i]; bi = i; }
        rv[threadIdx.x] = best; ri[threadIdx.x] = bi;
        __syncthreads();
        for (int o = 128; o > 0; o >>= 1) {
            if (threadIdx.x < o) {
                const float vo = rv[threadIdx.x + o];
                if (vo > rv[threadIdx.x] ||
                    (vo == rv[threadIdx.x] && ri[threadIdx.x + o] < ri[threadIdx.x])) {
                    rv[threadIdx.x] = vo; ri[threadIdx.x] = ri[threadIdx.x + o];
                }
            }
            __syncthreads();
        }
        if (threadIdx.x == 0) { tv[t] = rv[0]; tix[t] = ri[0]; vals[ri[0]] = -3.4e38f; }
        __syncthreads();
    }
    if (threadIdx.x == 0) {
        float mx = tv[0];
        for (int t = 1; t < topk; ++t) mx = tv[t] > mx ? tv[t] : mx;
        float se = 0.f;
        for (int t = 0; t < topk; ++t) { const float e = __expf(tv[t] - mx); tv[t] = e; se += e; }
        const float inv = 1.0f / se;
        for (int t = 0; t < topk; ++t) { wts[b * 32 + t] = tv[t] * inv; delays[b * 32 + t] = tix[t]; }
    }
}

// out[b,l,d] = sum_t w[b,t] * v[b,(l+delay[b,t])%L,d]
__global__ void agg_kernel(const float* __restrict__ v,
                           const int* __restrict__ delays,
                           const float* __restrict__ wts,
                           float* __restrict__ out, int L, int topk)
{
    const size_t n = (size_t)BATCH * L * D_MODEL;
    const size_t idx = (size_t)blockIdx.x * blockDim.x + threadIdx.x;
    if (idx >= n) return;
    const int c = (int)(idx % D_MODEL);
    const size_t bl = idx / D_MODEL;
    const int l = (int)(bl % L);
    const int b = (int)(bl / L);
    const float* vb = v + (size_t)b * L * D_MODEL + c;
    float s = 0.f;
    for (int t = 0; t < topk; ++t) {
        int ls = l + delays[b * 32 + t];
        if (ls >= L) ls -= L;
        s += wts[b * 32 + t] * vb[(size_t)ls * D_MODEL];
    }
    out[idx] = s;
}

// dst[B,Lq,D]: first S rows copied from src[B,S,D], rest zero
__global__ void pad_copy_kernel(const float* __restrict__ src,
                                float* __restrict__ dst, int S, int Lq)
{
    const size_t n = (size_t)BATCH * Lq * D_MODEL;
    const size_t idx = (size_t)blockIdx.x * blockDim.x + threadIdx.x;
    if (idx >= n) return;
    const int c = (int)(idx % D_MODEL);
    const size_t bl = idx / D_MODEL;
    const int l = (int)(bl % Lq);
    const int b = (int)(bl / Lq);
    dst[idx] = (l < S) ? src[((size_t)b * S + l) * D_MODEL + c] : 0.0f;
}

// decoder inits: seasonal_init / trend_init from encoder decomposition tail
__global__ void decinit_kernel(const float* __restrict__ seas_enc,
                               const float* __restrict__ trend_enc,
                               const float* __restrict__ mean_bc,
                               float* __restrict__ seas_init,
                               float* __restrict__ trend_out)
{
    const size_t n = (size_t)BATCH * L_DEC * D_MODEL;
    const size_t idx = (size_t)blockIdx.x * blockDim.x + threadIdx.x;
    if (idx >= n) return;
    const int c = (int)(idx % D_MODEL);
    const size_t bl = idx / D_MODEL;
    const int l = (int)(bl % L_DEC);
    const int b = (int)(bl / L_DEC);
    if (l < LABEL) {
        const size_t src = ((size_t)b * L_ENC + (L_ENC - LABEL + l)) * D_MODEL + c;
        seas_init[idx] = seas_enc[src];
        trend_out[idx] = trend_enc[src];
    } else {
        seas_init[idx] = 0.0f;
        trend_out[idx] = mean_bc[b * D_MODEL + c];
    }
}

__global__ void zero_kernel(float* __restrict__ p, size_t n)
{
    const size_t idx = (size_t)blockIdx.x * blockDim.x + threadIdx.x;
    if (idx < n) p[idx] = 0.0f;
}

// out[b,l] = trend[b,l,511] + proj_b[511] + x[b,l,:] . projW[511,:]
__global__ void final_kernel(const float* __restrict__ x,
                             const float* __restrict__ Wp,
                             const float* __restrict__ bp,
                             const float* __restrict__ trend,
                             float* __restrict__ out)
{
    const int idx = blockIdx.x * blockDim.x + threadIdx.x;
    if (idx >= BATCH * L_DEC) return;
    const float* xr = x + (size_t)idx * D_MODEL;
    const float* wr = Wp + (size_t)(D_MODEL - 1) * D_MODEL;
    float s = bp[D_MODEL - 1];
    for (int d = 0; d < D_MODEL; ++d) s += xr[d] * wr[d];
    out[idx] = s + trend[(size_t)idx * D_MODEL + (D_MODEL - 1)];
}

// ---------------------------------------------------------------------------
// Host-side orchestration
// ---------------------------------------------------------------------------
struct AttnParams { const float *Wq,*bq,*Wk,*bk,*Wv,*bv,*Wo,*bo; };

static void pack_bf16(hipStream_t st, const float* src, __bf16* dst, size_t n)
{
    pack_bf16_kernel<<<(unsigned)((n + 255) / 256), 256, 0, st>>>(src, dst, n);
}

// pack W (fp32 [N,K]) then run GEMM; output fp32 (Cf) or bf16 (Cb)
static void gemm_full(hipStream_t st, const __bf16* A, const float* W, __bf16* Wscr,
                      const float* bias, const float* res, float* Cf, __bf16* Cb,
                      int M, int N, int K, int act)
{
    pack_bf16(st, W, Wscr, (size_t)N * K);
    dim3 blk(32, 4);
    dim3 grd(N / (16 * TN), (M / (16 * TM) + 3) / 4);
    gemm_bf16_wmma_kernel<<<grd, blk, 0, st>>>(A, Wscr, bias, res, Cf, Cb, M, N, K, act);
}

static void circconv_full(hipStream_t st, const __bf16* Xbf, const float* W, __bf16* Wscr,
                          const float* mark, const float* Wtemp, const float* res,
                          float* C, int L, int Cin, int O)
{
    const size_t nw = (size_t)3 * O * Cin;
    pack_convw_kernel<<<(unsigned)((nw + 255) / 256), 256, 0, st>>>(W, Wscr, O, Cin);
    dim3 blk(32, 4);
    dim3 grd(O / 32, (BATCH * L / 16 + 3) / 4);
    circconv_bf16_wmma_kernel<<<grd, blk, 0, st>>>(Xbf, Wscr, mark, Wtemp, res, C, L, Cin, O);
}

static void run_attention(hipStream_t st, const AttnParams& p,
                          const float* xq, const float* xkv, int Lq, int S,
                          float* qb, float* kb, float* vb, float* tmpf,
                          __bf16* AbfQ, __bf16* AbfKV, __bf16* Wscr,
                          float* corr, int* delays, float* wts, float* outb)
{
    const size_t nq = (size_t)BATCH * Lq * D_MODEL;
    pack_bf16(st, xq, AbfQ, nq);
    gemm_full(st, AbfQ, p.Wq, Wscr, p.bq, nullptr, qb, nullptr,
              BATCH * Lq, D_MODEL, D_MODEL, 0);
    const __bf16* Akv = AbfQ;
    if (xkv != xq) {
        pack_bf16(st, xkv, AbfKV, (size_t)BATCH * S * D_MODEL);
        Akv = AbfKV;
    }
    if (S == Lq) {
        gemm_full(st, Akv, p.Wk, Wscr, p.bk, nullptr, kb, nullptr,
                  BATCH * S, D_MODEL, D_MODEL, 0);
        gemm_full(st, Akv, p.Wv, Wscr, p.bv, nullptr, vb, nullptr,
                  BATCH * S, D_MODEL, D_MODEL, 0);
    } else {
        gemm_full(st, Akv, p.Wk, Wscr, p.bk, nullptr, tmpf, nullptr,
                  BATCH * S, D_MODEL, D_MODEL, 0);
        pad_copy_kernel<<<(unsigned)((nq + 255) / 256), 256, 0, st>>>(tmpf, kb, S, Lq);
        gemm_full(st, Akv, p.Wv, Wscr, p.bv, nullptr, tmpf, nullptr,
                  BATCH * S, D_MODEL, D_MODEL, 0);
        pad_copy_kernel<<<(unsigned)((nq + 255) / 256), 256, 0, st>>>(tmpf, vb, S, Lq);
    }
    meancorr_kernel<<<dim3(Lq, BATCH), 256, 0, st>>>(qb, kb, corr, Lq);
    const int topk = (int)(3.0 * log((double)Lq));   // 18 (L=512) / 19 (L=768)
    topk_softmax_kernel<<<BATCH, 256, 0, st>>>(corr, delays, wts, Lq, topk);
    agg_kernel<<<(unsigned)((nq + 255) / 256), 256, 0, st>>>(vb, delays, wts, qb, Lq, topk);
    pack_bf16(st, qb, AbfQ, nq);                       // agg -> bf16
    gemm_full(st, AbfQ, p.Wo, Wscr, p.bo, /*res=*/xq, outb, nullptr,
              BATCH * Lq, D_MODEL, D_MODEL, 0);
}

extern "C" void kernel_launch(void* const* d_in, const int* in_sizes, int n_in,
                              void* d_out, int out_size, void* d_ws, size_t ws_size,
                              hipStream_t stream)
{
    (void)in_sizes; (void)n_in; (void)out_size; (void)ws_size;
    const float* x_enc      = (const float*)d_in[0];
    const float* x_mark_enc = (const float*)d_in[1];
    /* d_in[2] = x_dec: numerically unused by the reference (only its shape) */
    const float* x_mark_dec = (const float*)d_in[3];

    // Param leaves in JAX pytree order (dict keys sorted alphabetically):
    const float* dec_Wtemp = (const float*)d_in[4];
    const float* dec_Wtok  = (const float*)d_in[5];
    const float* dW1  = (const float*)d_in[6];
    const float* dW2  = (const float*)d_in[7];
    const float* dWtr = (const float*)d_in[8];
    AttnParams cross = { (const float*)d_in[11], (const float*)d_in[15],   // Wq,bq
                         (const float*)d_in[9],  (const float*)d_in[13],   // Wk,bk
                         (const float*)d_in[12], (const float*)d_in[16],   // Wv,bv
                         (const float*)d_in[10], (const float*)d_in[14] }; // Wo,bo
    AttnParams dself = { (const float*)d_in[19], (const float*)d_in[23],
                         (const float*)d_in[17], (const float*)d_in[21],
                         (const float*)d_in[20], (const float*)d_in[24],
                         (const float*)d_in[18], (const float*)d_in[22] };
    const float* dec_nb = (const float*)d_in[25];
    const float* dec_ng = (const float*)d_in[26];
    const float* enc_Wtemp = (const float*)d_in[27];
    const float* enc_Wtok  = (const float*)d_in[28];
    const float* eW1[2]; const float* eW2[2]; AttnParams eattn[2];
    for (int li = 0; li < 2; ++li) {
        const int base = 29 + li * 10;
        eW1[li] = (const float*)d_in[base + 0];
        eW2[li] = (const float*)d_in[base + 1];
        eattn[li] = { (const float*)d_in[base + 4], (const float*)d_in[base + 8],
                      (const float*)d_in[base + 2], (const float*)d_in[base + 6],
                      (const float*)d_in[base + 5], (const float*)d_in[base + 9],
                      (const float*)d_in[base + 3], (const float*)d_in[base + 7] };
    }
    const float* enc_nb = (const float*)d_in[49];
    const float* enc_ng = (const float*)d_in[50];
    const float* projW  = (const float*)d_in[51];
    const float* projb  = (const float*)d_in[52];
    float* out = (float*)d_out;

    // Workspace (bump allocated; fp32 units)
    float* ws = (float*)d_ws;
    size_t off = 0;
    auto alloc = [&](size_t nf) { float* p = ws + off; off += nf; return p; };
    const size_t TD = (size_t)BATCH * L_DEC * D_MODEL;     // 12.58M elems
    const size_t TE = (size_t)BATCH * L_ENC * D_MODEL;     //  8.39M elems
    float*  xbuf   = alloc(TD);
    float*  Bq     = alloc(TD);
    float*  Bk     = alloc(TD);
    float*  Bv     = alloc(TD);
    float*  trend  = alloc(TD);
    float*  tsum   = alloc(TD);
    float*  encout = alloc(TE);
    float*  tmpf   = alloc(TE);                            // cross-attn k/v pre-pad
    __bf16* Abf1   = (__bf16*)alloc(((size_t)BATCH * L_DEC * D_FF) / 2);  // bf16 hidden
    __bf16* Abf2   = (__bf16*)alloc(TD / 2);               // bf16 activations
    __bf16* Wbf    = (__bf16*)alloc((size_t)512 * 1024);   // 1M bf16 weight scratch
    float*  corr   = alloc(BATCH * L_DEC);
    float*  meanbc = alloc(BATCH * D_MODEL);
    float*  wts    = alloc(BATCH * 32);
    int*    delays = (int*)alloc(BATCH * 32);

    const unsigned GE = (unsigned)((TE + 255) / 256);
    const unsigned GD = (unsigned)((TD + 255) / 256);

    // ---------------- Encoder ----------------
    pack_bf16(stream, x_enc, Abf2, TE);
    circconv_full(stream, Abf2, enc_Wtok, Wbf, x_mark_enc, enc_Wtemp, nullptr,
                  xbuf, L_ENC, D_MODEL, D_MODEL);
    for (int li = 0; li < 2; ++li) {
        run_attention(stream, eattn[li], xbuf, xbuf, L_ENC, L_ENC,
                      Bq, Bk, Bv, tmpf, Abf2, Abf2, Wbf, corr, delays, wts, /*out*/Bk);
        decomp_kernel<<<GE, 256, 0, stream>>>(Bk, xbuf, nullptr, L_ENC, D_MODEL, 0);
        pack_bf16(stream, xbuf, Abf2, TE);
        gemm_full(stream, Abf2, eW1[li], Wbf, nullptr, nullptr, nullptr, /*Cb*/Abf1,
                  BATCH * L_ENC, D_FF, D_MODEL, /*gelu*/1);
        gemm_full(stream, Abf1, eW2[li], Wbf, nullptr, /*res*/xbuf, Bk, nullptr,
                  BATCH * L_ENC, D_MODEL, D_FF, 0);
        decomp_kernel<<<GE, 256, 0, stream>>>(Bk, xbuf, nullptr, L_ENC, D_MODEL, 0);
    }
    layernorm_kernel<<<BATCH * L_ENC, 256, 0, stream>>>(xbuf, enc_ng, enc_nb, encout, D_MODEL);
    subtimemean_kernel<<<(BATCH * D_MODEL + 255) / 256, 256, 0, stream>>>(encout, L_ENC, D_MODEL);

    // ---------------- Decoder init ----------------
    meantime_kernel<<<(BATCH * D_MODEL + 255) / 256, 256, 0, stream>>>(x_enc, meanbc, L_ENC, D_MODEL);
    decomp_kernel<<<GE, 256, 0, stream>>>(x_enc, Bq, Bk, L_ENC, D_MODEL, 0);
    decinit_kernel<<<GD, 256, 0, stream>>>(Bq, Bk, meanbc, /*seas_init*/Bv, trend);
    pack_bf16(stream, Bv, Abf2, TD);
    circconv_full(stream, Abf2, dec_Wtok, Wbf, x_mark_dec, dec_Wtemp, nullptr,
                  xbuf, L_DEC, D_MODEL, D_MODEL);
    zero_kernel<<<GD, 256, 0, stream>>>(tsum, TD);

    // ---------------- Decoder layer ----------------
    run_attention(stream, dself, xbuf, xbuf, L_DEC, L_DEC,
                  Bq, Bk, Bv, tmpf, Abf2, Abf2, Wbf, corr, delays, wts, Bk);
    decomp_kernel<<<GD, 256, 0, stream>>>(Bk, xbuf, tsum, L_DEC, D_MODEL, 1);
    run_attention(stream, cross, xbuf, encout, L_DEC, L_ENC,
                  Bq, Bk, Bv, tmpf, Abf2, (__bf16*)Abf1, Wbf, corr, delays, wts, Bk);
    decomp_kernel<<<GD, 256, 0, stream>>>(Bk, xbuf, tsum, L_DEC, D_MODEL, 1);
    pack_bf16(stream, xbuf, Abf2, TD);
    gemm_full(stream, Abf2, dW1, Wbf, nullptr, nullptr, nullptr, /*Cb*/Abf1,
              BATCH * L_DEC, D_FF, D_MODEL, 1);
    gemm_full(stream, Abf1, dW2, Wbf, nullptr, /*res*/xbuf, Bk, nullptr,
              BATCH * L_DEC, D_MODEL, D_FF, 0);
    decomp_kernel<<<GD, 256, 0, stream>>>(Bk, xbuf, tsum, L_DEC, D_MODEL, 1);
    pack_bf16(stream, tsum, Abf2, TD);
    circconv_full(stream, Abf2, dWtr, Wbf, nullptr, nullptr, /*res*/trend, trend,
                  L_DEC, D_MODEL, D_MODEL);

    // ---------------- Head ----------------
    layernorm_kernel<<<BATCH * L_DEC, 256, 0, stream>>>(xbuf, dec_ng, dec_nb, Bq, D_MODEL);
    subtimemean_kernel<<<(BATCH * D_MODEL + 255) / 256, 256, 0, stream>>>(Bq, L_DEC, D_MODEL);
    final_kernel<<<(BATCH * L_DEC + 255) / 256, 256, 0, stream>>>(Bq, projW, projb, trend, out);
}
